// WindowAttention_14018773254205
// MI455X (gfx1250) — matrix operand order repaired and checked by reading, not compiled, not run on hardware
//
#include <hip/hip_runtime.h>
#include <hip/hip_bf16.h>

typedef _Float16 v8h  __attribute__((ext_vector_type(8)));
typedef _Float16 v16h __attribute__((ext_vector_type(16)));
typedef float    v8f  __attribute__((ext_vector_type(8)));

// Assemble a 16-half WMMA fragment from two 16-byte (8-half) aligned chunks.
__device__ __forceinline__ v16h mk_frag(const _Float16* p0, const _Float16* p1) {
    v8h lo = *reinterpret_cast<const v8h*>(p0);
    v8h hi = *reinterpret_cast<const v8h*>(p1);
    return __builtin_shufflevector(lo, hi, 0,1,2,3,4,5,6,7,8,9,10,11,12,13,14,15);
}

#define WMMA_F16(A, B, C) \
    __builtin_amdgcn_wmma_f32_16x16x32_f16(false, (A), false, (B), (short)0, (C), false, false)

// ---------------------------------------------------------------------------
// Prep kernel: f32->f16 weight conversion + relative-position bias gather.
// ws layout (bytes): [0) qkv_w f16 384KB | 393216) proj_w f16 128KB | 524288) bias f32 128KB
// ---------------------------------------------------------------------------
__global__ void winattn_prep(const float* __restrict__ qkv_w,
                             const float* __restrict__ proj_w,
                             const float* __restrict__ bias_table,
                             _Float16* __restrict__ qkvw16,
                             _Float16* __restrict__ projw16,
                             float* __restrict__ biasf) {
    int i = blockIdx.x * 256 + threadIdx.x;
    if (i < 768 * 256) qkvw16[i] = (_Float16)qkv_w[i];
    if (i < 256 * 256) projw16[i] = (_Float16)proj_w[i];
    if (i < 8 * 64 * 64) {
        int h = i >> 12, rem = i & 4095;
        int n = rem >> 6, m = rem & 63;
        int r0 = (n >> 3) - (m >> 3) + 7;   // dh + 7
        int r1 = (n & 7)  - (m & 7)  + 7;   // dw + 7
        biasf[i] = bias_table[(r0 * 15 + r1) * 8 + h];
    }
}

// ---------------------------------------------------------------------------
// Fused window-attention kernel: one workgroup (8 wave32) per window.
// LDS (halfs):
//   qT   [64][520]  : q,k transposed -> row n, col d (q: 0..255, k: 256..511)
//   vs   [256][72]  : v as [d][m]
//   outb [64][264]  : attn@V output, row n, col h*32+d
//   xsT  [64][264]  : x transposed (row n, col c) -- dead after GEMM1, region
//                     reused for attnf f32 [64][68] + attn16 [64][72]
// ---------------------------------------------------------------------------
#define LDS_BYTES 171008

__global__ __launch_bounds__(256) void winattn_main(
        const float* __restrict__ x,        // [4096][256][64]
        const float* __restrict__ mask,     // [4096][64][64]
        const float* __restrict__ qkv_b,    // [768]
        const float* __restrict__ proj_b,   // [256]
        const _Float16* __restrict__ qkvw16,// [768][256]
        const _Float16* __restrict__ projw16,// [256][256]
        const float* __restrict__ biasf,    // [8][64][64]
        float* __restrict__ out) {          // [4096][64][256]
    extern __shared__ __align__(16) char smem[];
    _Float16* qT   = (_Float16*)smem;                 // 64*520
    _Float16* vs   = qT + 64 * 520;                   // 256*72
    _Float16* outb = vs + 256 * 72;                   // 64*264
    _Float16* xsT  = outb + 64 * 264;                 // 64*264 (scratch union)
    float*    attnf  = (float*)xsT;                   // 64*68 f32
    _Float16* attn16 = xsT + 64 * 68 * 2;             // 64*72 f16 (after attnf)

    const int tid   = threadIdx.x;
    const int w     = tid >> 5;
    const int lane  = tid & 31;
    const int lhalf = lane >> 4;   // which 16-lane half
    const int lm    = lane & 15;   // lane index within half
    const int b     = blockIdx.x;

    // ---- Phase 1: stage x transposed as f16: xsT[n][c] ----
    for (int i = tid; i < 256 * 64; i += 256) {
        int c = i >> 6, n = i & 63;
        xsT[n * 264 + c] = (_Float16)x[b * 16384 + i];
    }
    __syncthreads();

    // ---- Phase 2: QKV GEMM: qkv[d][n] = qkv_w[d][:] . x[:][n] + qkv_b[d] ----
    // 48 M-tiles (d) x 4 N-tiles (n); 24 tiles per wave; K = 256 (8 steps).
    for (int t = w; t < 192; t += 8) {
        int mt = t >> 2, nt = t & 3;
        int dbase = mt * 16;
        v8f acc;
        #pragma unroll
        for (int r = 0; r < 8; ++r) acc[r] = qkv_b[dbase + r + 8 * lhalf];
        int arow = dbase + lm;          // A lane: M = d
        int brow = nt * 16 + lm;        // B lane: N = n
        #pragma unroll
        for (int kk = 0; kk < 8; ++kk) {
            int K0 = kk * 32;
            v16h A = mk_frag(qkvw16 + arow * 256 + K0 + 8 * lhalf,
                             qkvw16 + arow * 256 + K0 + 16 + 8 * lhalf);
            v16h B = mk_frag(xsT + brow * 264 + K0 + 8 * lhalf,
                             xsT + brow * 264 + K0 + 16 + 8 * lhalf);
            acc = WMMA_F16(A, B, acc);
        }
        int n = nt * 16 + lm;           // D lane: N = n
        if (mt < 32) {                  // q,k -> transposed [n][d]
            #pragma unroll
            for (int r = 0; r < 8; ++r) {
                int d = dbase + r + 8 * lhalf;
                qT[n * 520 + d] = (_Float16)acc[r];
            }
        } else {                        // v -> [d][m]
            #pragma unroll
            for (int r = 0; r < 8; ++r) {
                int d = dbase - 512 + r + 8 * lhalf;
                vs[d * 72 + n] = (_Float16)acc[r];
            }
        }
    }
    __syncthreads();

    const float scale = 0.17677669529663687f;  // 32^-0.5

    // ---- Phase 3: per-head attention ----
    for (int h = 0; h < 8; ++h) {
        // attn[n][m] = scale * q[:,n].k[:,m] + bias[h][n][m] + mask[b][n][m]
        for (int t = w; t < 16; t += 8) {
            int mt = t >> 2, nt = t & 3;
            v8f acc;
            #pragma unroll
            for (int r = 0; r < 8; ++r) acc[r] = 0.0f;
            int nrow = mt * 16 + lm;    // A lane: M = n
            int mcol = nt * 16 + lm;    // B lane: N = m
            v16h A = mk_frag(qT + nrow * 520 + h * 32 + 8 * lhalf,
                             qT + nrow * 520 + h * 32 + 16 + 8 * lhalf);
            v16h B = mk_frag(qT + mcol * 520 + 256 + h * 32 + 8 * lhalf,
                             qT + mcol * 520 + 256 + h * 32 + 16 + 8 * lhalf);
            acc = WMMA_F16(A, B, acc);
            #pragma unroll
            for (int r = 0; r < 8; ++r) {
                int n = mt * 16 + r + 8 * lhalf;
                float v = acc[r] * scale
                        + biasf[h * 4096 + n * 64 + mcol]
                        + mask[b * 4096 + n * 64 + mcol];
                attnf[n * 68 + mcol] = v;
            }
        }
        __syncthreads();

        // softmax over m: 4 lanes per row, shuffle-reduce max & sum
        {
            int n = tid >> 2, seg = tid & 3;
            float vals[16];
            float mx = -3.4e38f;
            #pragma unroll
            for (int i = 0; i < 16; ++i) {
                vals[i] = attnf[n * 68 + seg * 16 + i];
                mx = fmaxf(mx, vals[i]);
            }
            mx = fmaxf(mx, __shfl_xor(mx, 1, 32));
            mx = fmaxf(mx, __shfl_xor(mx, 2, 32));
            float s = 0.f;
            #pragma unroll
            for (int i = 0; i < 16; ++i) { vals[i] = __expf(vals[i] - mx); s += vals[i]; }
            s += __shfl_xor(s, 1, 32);
            s += __shfl_xor(s, 2, 32);
            float inv = 1.0f / s;
            #pragma unroll
            for (int i = 0; i < 16; ++i)
                attn16[n * 72 + seg * 16 + i] = (_Float16)(vals[i] * inv);
        }
        __syncthreads();

        // out_h[n][d] = attn[n][:] . v[d][:]   (4 M-tiles x 2 N-tiles, 1/wave)
        {
            int mt = w >> 1, nt = w & 1;
            v8f acc;
            #pragma unroll
            for (int r = 0; r < 8; ++r) acc[r] = 0.0f;
            int nrow = mt * 16 + lm;    // A lane: M = n
            int dcol = nt * 16 + lm;    // B lane: N = d
            #pragma unroll
            for (int kk = 0; kk < 2; ++kk) {
                int K0 = kk * 32;
                v16h A = mk_frag(attn16 + nrow * 72 + K0 + 8 * lhalf,
                                 attn16 + nrow * 72 + K0 + 16 + 8 * lhalf);
                v16h B = mk_frag(vs + (h * 32 + dcol) * 72 + K0 + 8 * lhalf,
                                 vs + (h * 32 + dcol) * 72 + K0 + 16 + 8 * lhalf);
                acc = WMMA_F16(A, B, acc);
            }
            int cc = h * 32 + nt * 16 + lm;
            #pragma unroll
            for (int r = 0; r < 8; ++r) {
                int n = mt * 16 + r + 8 * lhalf;
                outb[n * 264 + cc] = (_Float16)acc[r];
            }
        }
        __syncthreads();
    }

    // ---- Phase 4: projection: final[n][dout] = outb[n][:] . proj_w[dout][:] + proj_b ----
    // 4 M-tiles x 16 N-tiles; 8 tiles per wave; K = 256 (8 steps).
    for (int t = w; t < 64; t += 8) {
        int mt = t >> 4, nt = t & 15;
        int dout = nt * 16 + lm;
        float pb = proj_b[dout];
        v8f acc;
        #pragma unroll
        for (int r = 0; r < 8; ++r) acc[r] = pb;
        int nrow = mt * 16 + lm;
        #pragma unroll
        for (int kk = 0; kk < 8; ++kk) {
            int K0 = kk * 32;
            v16h A = mk_frag(outb + nrow * 264 + K0 + 8 * lhalf,
                             outb + nrow * 264 + K0 + 16 + 8 * lhalf);
            v16h B = mk_frag(projw16 + dout * 256 + K0 + 8 * lhalf,
                             projw16 + dout * 256 + K0 + 16 + 8 * lhalf);
            acc = WMMA_F16(A, B, acc);
        }
        #pragma unroll
        for (int r = 0; r < 8; ++r) {
            int n = mt * 16 + r + 8 * lhalf;
            out[b * 16384 + n * 256 + dout] = acc[r];
        }
    }
}

extern "C" void kernel_launch(void* const* d_in, const int* in_sizes, int n_in,
                              void* d_out, int out_size, void* d_ws, size_t ws_size,
                              hipStream_t stream) {
    (void)in_sizes; (void)n_in; (void)out_size; (void)ws_size;
    const float* x          = (const float*)d_in[0];
    const float* mask       = (const float*)d_in[1];
    const float* qkv_w      = (const float*)d_in[2];
    const float* qkv_b      = (const float*)d_in[3];
    const float* proj_w     = (const float*)d_in[4];
    const float* proj_b     = (const float*)d_in[5];
    const float* bias_table = (const float*)d_in[6];

    char* ws = (char*)d_ws;
    _Float16* qkvw16 = (_Float16*)ws;                   // 768*256*2 = 393216 B
    _Float16* projw16 = (_Float16*)(ws + 393216);       // 256*256*2 = 131072 B
    float*    biasf   = (float*)(ws + 524288);          // 8*64*64*4 = 131072 B

    winattn_prep<<<768, 256, 0, stream>>>(qkv_w, proj_w, bias_table,
                                          qkvw16, projw16, biasf);

    (void)hipFuncSetAttribute(reinterpret_cast<const void*>(winattn_main),
                              hipFuncAttributeMaxDynamicSharedMemorySize, LDS_BYTES);
    winattn_main<<<4096, 256, LDS_BYTES, stream>>>(x, mask, qkv_b, proj_b,
                                                   qkvw16, projw16, biasf,
                                                   (float*)d_out);
}
